// SlotAttention_52510270161414
// MI455X (gfx1250) — matrix operand order repaired and checked by reading, not compiled, not run on hardware
//
#include <hip/hip_runtime.h>

// ---------------------------------------------------------------------------
// Slot Attention on MI455X (gfx1250).
//   - All matmuls via v_wmma_f32_16x16x32_bf16, 16x64 output per wave (4 acc).
//   - K/V projection uses the Tensor Data Mover (tensor_load_to_lds) to stage
//     the A slab in LDS, synchronized with s_wait_tensorcnt + barrier.
//   - fp32 master state for slots / softmax / GRU math.
// ---------------------------------------------------------------------------

typedef __attribute__((ext_vector_type(16))) __bf16 v16bf;
typedef __attribute__((ext_vector_type(8)))  float  v8f;

typedef unsigned int tdm_u32x4 __attribute__((ext_vector_type(4)));
typedef int          tdm_i32x4 __attribute__((ext_vector_type(4)));
typedef int          tdm_i32x8 __attribute__((ext_vector_type(8)));

#define F_OUTBF16 1   // store D as bf16 (else f32)
#define F_ROWS8   2   // only store local tile rows < 8 (slot-padded tiles)
#define F_RELU    8   // relu after bias

#if __has_builtin(__builtin_amdgcn_tensor_load_to_lds)
#define USE_TDM 1
#endif

// ---- WMMA tile loads -------------------------------------------------------
// A operand (16x32 bf16, row-major [M][K] source):
//   lane<16 : row=lane,    halves 0..7 = K 0..7,  halves 8..15 = K 16..23
//   lane>=16: row=lane-16, halves 0..7 = K 8..15, halves 8..15 = K 24..31
__device__ __forceinline__ v16bf load_tile_a(const __bf16* p0, int lda) {
  int lane = threadIdx.x & 31;
  const __bf16* p = p0 + (lane & 15) * lda + ((lane >> 4) << 3);
  union { v16bf v; uint4 q[2]; } t;
  t.q[0] = *reinterpret_cast<const uint4*>(p);
  t.q[1] = *reinterpret_cast<const uint4*>(p + 16);
  return t.v;
}

// B operand (32x16 bf16, [K][N] source): lane l holds N=l%16; halves 0..7 = K 8h..8h+7, halves 8..15 = K 16+8h..16+8h+7 (h=l/16).
__device__ __forceinline__ v16bf load_tile_b(const __bf16* p0, int ldb) {
  int lane = threadIdx.x & 31;
  const __bf16* p = p0 + (lane & 15);
  int kb = (lane >> 4) << 3;
  v16bf t;
#pragma unroll
  for (int s = 0; s < 16; ++s) t[s] = p[(long long)(kb + (s < 8 ? s : s + 8)) * ldb];
  return t;
}

__device__ __forceinline__ v8f wmma_bf16(v16bf a, v16bf b, v8f c) {
  return __builtin_amdgcn_wmma_f32_16x16x32_bf16(false, a, false, b,
                                                 (short)0, c, false, false);
}

// ---- Generic batched WMMA GEMM: one wave per 16x64 output block ------------
__global__ void __launch_bounds__(256) wmma_gemm_kernel(
    const __bf16* __restrict__ A, const __bf16* __restrict__ B,
    void* __restrict__ C,
    int Mt, int Nt4, int K, int lda, int ldb, int ldc,
    long long sA, long long sB, long long sC,
    const float* __restrict__ bias, const float* __restrict__ residual,
    float alpha, int flags)
{
  int tile = (int)((blockIdx.x * blockDim.x + threadIdx.x) >> 5);
  if (tile >= Mt * Nt4) return;                 // whole-wave uniform exit
  int tm = tile / Nt4;
  int n0 = (tile % Nt4) * 64;
  long long batch = blockIdx.y;
  const __bf16* Ab = A + batch * sA + (long long)tm * 16 * lda;
  const __bf16* Bb = B + batch * sB + n0;

  v8f acc0 = {}, acc1 = {}, acc2 = {}, acc3 = {};
  for (int k0 = 0; k0 < K; k0 += 32) {
    v16bf a = load_tile_a(Ab + k0, lda);
    const __bf16* bp = Bb + (long long)k0 * ldb;
    v16bf b0 = load_tile_b(bp,      ldb);
    v16bf b1 = load_tile_b(bp + 16, ldb);
    v16bf b2 = load_tile_b(bp + 32, ldb);
    v16bf b3 = load_tile_b(bp + 48, ldb);
    acc0 = wmma_bf16(a, b0, acc0);
    acc1 = wmma_bf16(a, b1, acc1);
    acc2 = wmma_bf16(a, b2, acc2);
    acc3 = wmma_bf16(a, b3, acc3);
  }

  // D layout: VGPR r, lane<16 -> (M=r, N=lane); lane>=16 -> (M=r+8, N=lane-16)
  int lane  = threadIdx.x & 31;
  int col   = lane & 15;
  int mhalf = (lane >> 4) << 3;
  v8f accs[4] = {acc0, acc1, acc2, acc3};
#pragma unroll
  for (int t = 0; t < 4; ++t) {
    int gn = n0 + t * 16 + col;
    float badd = bias ? bias[gn] : 0.0f;
#pragma unroll
    for (int r = 0; r < 8; ++r) {
      int lm = r + mhalf;
      if ((flags & F_ROWS8) && lm >= 8) break;  // upper-half lanes drop rows 8..15
      int gm = tm * 16 + lm;
      float v = accs[t][r] * alpha + badd;
      if (flags & F_RELU) v = fmaxf(v, 0.0f);
      if (residual) v += residual[(long long)gm * ldc + gn];
      long long cidx = batch * sC + (long long)gm * ldc + gn;
      if (flags & F_OUTBF16) ((__bf16*)C)[cidx] = (__bf16)v;
      else                   ((float*)C)[cidx]  = v;
    }
  }
}

// ---- K/V projection with TDM-staged A slab ---------------------------------
// Workgroup = 8 waves, computes a 32-row x 256-col block of x @ W^T.
// Wave w: rows (w>>2)*16..+16, cols (w&3)*64..+64.  A slab (32x256 bf16,
// 16 KB) is DMA'd into LDS by the Tensor Data Mover, then read as ds_load.
__global__ void __launch_bounds__(256) proj_gemm_kernel(
    const __bf16* __restrict__ A,   // x_bf  [32768][256]
    const __bf16* __restrict__ B,   // W^T   [256][256] ([K][N])
    __bf16* __restrict__ C,
    int trans)                      // 1: C[b][d][j] (k^T)   0: C[b][j][d] (v)
{
  __shared__ __bf16 lds_a[32 * 256];
  const __bf16* slab = A + (unsigned long long)blockIdx.x * (32 * 256);

#ifdef USE_TDM
  if (threadIdx.x < 32) {           // wave 0 issues one TDM load (EXEC ignored)
    unsigned long long ga = (unsigned long long)(uintptr_t)slab;
    unsigned int laddr = (unsigned int)(uintptr_t)&lds_a[0];
    // D# group 0: count=1, lds_addr, 57-bit global_addr, type=2 ("image")
    tdm_u32x4 g0 = { 1u, laddr, (unsigned int)ga,
                     (unsigned int)((ga >> 32) & 0x01FFFFFFu) | (2u << 30) };
    // D# group 1: data_size=2B, tensor 256 x 32768, tile 256 x 32, stride 256
    tdm_i32x8 g1 = { (int)(1u << 16),        // [17:16] data_size = 1 (2 bytes)
                     (int)(256u << 16),      // [63:48] tensor_dim0 = 256
                     (int)(32768u << 16),    // [95:80] tensor_dim1 = 32768
                     (int)(256u << 16),      // [127:112] tile_dim0 = 256
                     32,                     // [143:128] tile_dim1 = 32
                     256,                    // [191:160] tensor_dim0_stride lo
                     0, 0 };
    tdm_i32x4 z4 = { 0, 0, 0, 0 };
#if __clang_major__ >= 23
    tdm_i32x8 z8 = { 0, 0, 0, 0, 0, 0, 0, 0 };
    __builtin_amdgcn_tensor_load_to_lds(g0, g1, z4, z4, z8, 0);
#else
    __builtin_amdgcn_tensor_load_to_lds(g0, g1, z4, z4, 0);
#endif
#if __has_builtin(__builtin_amdgcn_s_wait_tensorcnt)
    __builtin_amdgcn_s_wait_tensorcnt(0);
#else
    asm volatile("s_wait_tensorcnt 0x0" ::: "memory");
#endif
  }
  __syncthreads();
#else
  {
    const uint4* src = reinterpret_cast<const uint4*>(slab);
    uint4* dst = reinterpret_cast<uint4*>(lds_a);
    for (int i = threadIdx.x; i < 32 * 256 / 8; i += 256) dst[i] = src[i];
  }
  __syncthreads();
#endif

  int wave = threadIdx.x >> 5;
  int mt = wave >> 2;                 // 0..1
  int n0 = (wave & 3) * 64;           // 0,64,128,192
  const __bf16* Albase = lds_a + mt * 16 * 256;

  v8f acc0 = {}, acc1 = {}, acc2 = {}, acc3 = {};
#pragma unroll
  for (int k0 = 0; k0 < 256; k0 += 32) {
    v16bf a = load_tile_a(Albase + k0, 256);        // from LDS
    const __bf16* bp = B + (long long)k0 * 256 + n0;
    v16bf b0 = load_tile_b(bp,      256);
    v16bf b1 = load_tile_b(bp + 16, 256);
    v16bf b2 = load_tile_b(bp + 32, 256);
    v16bf b3 = load_tile_b(bp + 48, 256);
    acc0 = wmma_bf16(a, b0, acc0);
    acc1 = wmma_bf16(a, b1, acc1);
    acc2 = wmma_bf16(a, b2, acc2);
    acc3 = wmma_bf16(a, b3, acc3);
  }

  int lane  = threadIdx.x & 31;
  int col   = lane & 15;
  int mhalf = (lane >> 4) << 3;
  v8f accs[4] = {acc0, acc1, acc2, acc3};
#pragma unroll
  for (int t = 0; t < 4; ++t) {
    int gn = n0 + t * 16 + col;
#pragma unroll
    for (int r = 0; r < 8; ++r) {
      int gmg = blockIdx.x * 32 + mt * 16 + r + mhalf;  // global row
      int b = gmg >> 10, j = gmg & 1023;
      float v = accs[t][r];
      long long cidx = trans
          ? (long long)b * 262144 + (long long)gn * 1024 + j
          : (long long)gmg * 256 + gn;
      C[cidx] = (__bf16)v;
    }
  }
}

// ---- block-wide sum over 256 threads (8 wave32) ---------------------------
__device__ __forceinline__ float block_sum256(float v, float* sm) {
#pragma unroll
  for (int o = 16; o; o >>= 1) v += __shfl_xor(v, o, 32);
  int lane = threadIdx.x & 31;
  int w = threadIdx.x >> 5;
  if (lane == 0) sm[w] = v;
  __syncthreads();
  if (w == 0) {
    float t = (lane < 8) ? sm[lane] : 0.0f;
#pragma unroll
    for (int o = 4; o; o >>= 1) t += __shfl_xor(t, o, 32);
    if (lane == 0) sm[8] = t;
  }
  __syncthreads();
  float r = sm[8];
  __syncthreads();
  return r;
}

// ---- LayerNorm over a 256-wide row -> bf16 --------------------------------
__global__ void __launch_bounds__(256) ln_rows_kernel(
    const float* __restrict__ x, const float* __restrict__ g,
    const float* __restrict__ bta, __bf16* __restrict__ out)
{
  __shared__ float sm[12];
  long long row = blockIdx.x;
  int tid = threadIdx.x;
  float v = x[row * 256 + tid];
  float m = block_sum256(v, sm) * (1.0f / 256.0f);
  float dv = v - m;
  float var = block_sum256(dv * dv, sm) * (1.0f / 256.0f);
  float rs = rsqrtf(var + 1e-5f);
  out[row * 256 + tid] = (__bf16)(dv * rs * g[tid] + bta[tid]);
}

// ---- slots init: mu + exp(log_sigma)*noise --------------------------------
__global__ void __launch_bounds__(256) init_slots_kernel(
    const float* __restrict__ noise, const float* __restrict__ mu,
    const float* __restrict__ ls, float* __restrict__ slots)
{
  int t = blockIdx.x * 256 + threadIdx.x;
  int d = t & 255;
  slots[t] = mu[d] + __expf(ls[d]) * noise[t];
}

// ---- f32 -> bf16 (flat) ---------------------------------------------------
__global__ void __launch_bounds__(256) cvt_bf16_kernel(
    const float* __restrict__ src, __bf16* __restrict__ dst, int n)
{
  int t = blockIdx.x * 256 + threadIdx.x;
  if (t < n) dst[t] = (__bf16)src[t];
}

// ---- f32 [rows][cols] -> bf16 transposed [cols][rows] ---------------------
__global__ void __launch_bounds__(256) cvt_bf16_T_kernel(
    const float* __restrict__ src, __bf16* __restrict__ dst, int rows, int cols)
{
  int t = blockIdx.x * 256 + threadIdx.x;
  if (t >= rows * cols) return;
  int r = t / cols, c = t % cols;
  dst[c * rows + r] = (__bf16)src[t];
}

// ---- softmax over the 8 slots (axis=1) per (b, j) column ------------------
__global__ void __launch_bounds__(256) softmax_slots_kernel(
    const float* __restrict__ dots, float* __restrict__ attn)
{
  int t = blockIdx.x * 256 + threadIdx.x;   // 32768 threads: b*1024 + j
  int b = t >> 10, j = t & 1023;
  const float* p = dots + b * 8192 + j;
  float x[8];
  float mx = -3.0e38f;
#pragma unroll
  for (int i = 0; i < 8; ++i) { x[i] = p[i * 1024]; mx = fmaxf(mx, x[i]); }
  float s = 0.0f;
#pragma unroll
  for (int i = 0; i < 8; ++i) { x[i] = __expf(x[i] - mx); s += x[i]; }
  float inv = 1.0f / s;
  float* o = attn + b * 8192 + j;
#pragma unroll
  for (int i = 0; i < 8; ++i) o[i * 1024] = x[i] * inv + 1e-8f;
}

// ---- renormalize attn over j (1024) and emit bf16 -------------------------
__global__ void __launch_bounds__(256) attn_norm_kernel(
    const float* __restrict__ attn, __bf16* __restrict__ out)
{
  __shared__ float sm[12];
  long long row = blockIdx.x;               // b*8 + i, 256 rows
  int tid = threadIdx.x;
  const float* p = attn + row * 1024;
  float v0 = p[tid], v1 = p[tid + 256], v2 = p[tid + 512], v3 = p[tid + 768];
  float tot = block_sum256(v0 + v1 + v2 + v3, sm);
  float inv = 1.0f / tot;
  __bf16* o = out + row * 1024;
  o[tid]       = (__bf16)(v0 * inv);
  o[tid + 256] = (__bf16)(v1 * inv);
  o[tid + 512] = (__bf16)(v2 * inv);
  o[tid + 768] = (__bf16)(v3 * inv);
}

// ---- GRU cell elementwise + LayerNorm(ff) fused ---------------------------
__global__ void __launch_bounds__(256) gru_ln_kernel(
    const float* __restrict__ gi, const float* __restrict__ gh,
    const float* __restrict__ hprev,
    const float* __restrict__ g, const float* __restrict__ bta,
    float* __restrict__ slots_mid, __bf16* __restrict__ ffin)
{
  __shared__ float sm[12];
  long long row = blockIdx.x;
  int d = threadIdx.x;
  long long base = row * 768;
  float i_r = gi[base + d], i_z = gi[base + 256 + d], i_n = gi[base + 512 + d];
  float h_r = gh[base + d], h_z = gh[base + 256 + d], h_n = gh[base + 512 + d];
  float h = hprev[row * 256 + d];
  float r = 1.0f / (1.0f + __expf(-(i_r + h_r)));
  float z = 1.0f / (1.0f + __expf(-(i_z + h_z)));
  float nn = tanhf(i_n + r * h_n);
  float val = (1.0f - z) * nn + z * h;
  slots_mid[row * 256 + d] = val;
  float m = block_sum256(val, sm) * (1.0f / 256.0f);
  float dv = val - m;
  float var = block_sum256(dv * dv, sm) * (1.0f / 256.0f);
  float rs = rsqrtf(var + 1e-5f);
  ffin[row * 256 + d] = (__bf16)(dv * rs * g[d] + bta[d]);
}

__global__ void __launch_bounds__(256) copy_f32_kernel(
    const float* __restrict__ src, float* __restrict__ dst, int n)
{
  int t = blockIdx.x * 256 + threadIdx.x;
  if (t < n) dst[t] = src[t];
}

// ---------------------------------------------------------------------------
extern "C" void kernel_launch(void* const* d_in, const int* in_sizes, int n_in,
                              void* d_out, int out_size, void* d_ws, size_t ws_size,
                              hipStream_t stream) {
  (void)in_sizes; (void)n_in; (void)out_size; (void)ws_size;
  const float* inputs   = (const float*)d_in[0];   // [32,1024,256]
  const float* noise    = (const float*)d_in[1];   // [32,8,256]
  const float* slots_mu = (const float*)d_in[2];
  const float* slots_ls = (const float*)d_in[3];
  const float* Wq       = (const float*)d_in[4];   // [256,256]
  const float* Wk       = (const float*)d_in[5];
  const float* Wv       = (const float*)d_in[6];
  const float* gru_wih  = (const float*)d_in[7];   // [768,256]
  const float* gru_whh  = (const float*)d_in[8];
  const float* gru_bih  = (const float*)d_in[9];
  const float* gru_bhh  = (const float*)d_in[10];
  const float* mlp_w1   = (const float*)d_in[11];  // [1024,256]
  const float* mlp_b1   = (const float*)d_in[12];
  const float* mlp_w2   = (const float*)d_in[13];  // [256,1024]
  const float* mlp_b2   = (const float*)d_in[14];
  const float* ln_in_g  = (const float*)d_in[15];
  const float* ln_in_b  = (const float*)d_in[16];
  const float* ln_s_g   = (const float*)d_in[17];
  const float* ln_s_b   = (const float*)d_in[18];
  const float* ln_ff_g  = (const float*)d_in[19];
  const float* ln_ff_b  = (const float*)d_in[20];

  char* ws = (char*)d_ws;
  size_t off = 0;
  auto alloc = [&](size_t bytes) -> void* {
    off = (off + 255) & ~(size_t)255;
    void* p = ws + off;
    off += bytes;
    return p;
  };

  __bf16* x_bf   = (__bf16*)alloc(32768ull * 256 * 2);     // LN(inputs)
  __bf16* ktr    = (__bf16*)alloc(32ull * 256 * 1024 * 2); // k^T per batch [d][j]
  __bf16* vmat   = (__bf16*)alloc(32ull * 1024 * 256 * 2); // v per batch [j][d]
  __bf16* wqT    = (__bf16*)alloc(65536 * 2);
  __bf16* wkT    = (__bf16*)alloc(65536 * 2);
  __bf16* wvT    = (__bf16*)alloc(65536 * 2);
  __bf16* wihT   = (__bf16*)alloc(196608 * 2);   // [256][768]
  __bf16* whhT   = (__bf16*)alloc(196608 * 2);
  __bf16* w1T    = (__bf16*)alloc(262144 * 2);   // [256][1024]
  __bf16* w2T    = (__bf16*)alloc(262144 * 2);   // [1024][256]
  float*  slots  = (float*) alloc(65536 * 4);    // fp32 master
  float*  smid   = (float*) alloc(65536 * 4);
  __bf16* s_bf   = (__bf16*)alloc(264 * 256 * 2);  // LN(slots)/ff_in (+pad rows)
  __bf16* h_bf   = (__bf16*)alloc(65536 * 2);      // bf16(slots_prev)
  __bf16* q_bf   = (__bf16*)alloc(264 * 256 * 2);  // (+pad rows)
  float*  dots   = (float*) alloc(262144 * 4);     // [32][8][1024]
  float*  attn_f = (float*) alloc(262144 * 4);
  __bf16* attn_b = (__bf16*)alloc(264 * 1024 * 2); // (+pad rows)
  __bf16* u_bf   = (__bf16*)alloc(65536 * 2);
  float*  gi     = (float*) alloc(196608ull * 4);  // [256][768]
  float*  gh     = (float*) alloc(196608ull * 4);
  __bf16* hmid   = (__bf16*)alloc(262144 * 2);     // [256][1024]

  auto gemm = [&](const __bf16* A, const __bf16* B, void* C,
                  int Mt, int Nt, int K, int lda, int ldb, int ldc,
                  long long sA, long long sB, long long sC, int batches,
                  const float* bias, const float* residual,
                  float alpha, int flags) {
    int tiles = Mt * (Nt / 4);      // each wave covers 4 N-tiles (16x64)
    dim3 grid((tiles + 7) / 8, batches);
    wmma_gemm_kernel<<<grid, 256, 0, stream>>>(A, B, C, Mt, Nt / 4, K,
                                               lda, ldb, ldc, sA, sB, sC,
                                               bias, residual, alpha, flags);
  };

  // ---- one-time prep ----
  cvt_bf16_T_kernel<<<256,  256, 0, stream>>>(Wq, wqT, 256, 256);
  cvt_bf16_T_kernel<<<256,  256, 0, stream>>>(Wk, wkT, 256, 256);
  cvt_bf16_T_kernel<<<256,  256, 0, stream>>>(Wv, wvT, 256, 256);
  cvt_bf16_T_kernel<<<768,  256, 0, stream>>>(gru_wih, wihT, 768, 256);
  cvt_bf16_T_kernel<<<768,  256, 0, stream>>>(gru_whh, whhT, 768, 256);
  cvt_bf16_T_kernel<<<1024, 256, 0, stream>>>(mlp_w1, w1T, 1024, 256);
  cvt_bf16_T_kernel<<<1024, 256, 0, stream>>>(mlp_w2, w2T, 256, 1024);
  init_slots_kernel<<<256, 256, 0, stream>>>(noise, slots_mu, slots_ls, slots);
  ln_rows_kernel<<<32768, 256, 0, stream>>>(inputs, ln_in_g, ln_in_b, x_bf);

  // k^T[b][d][j] and v[b][j][d] via TDM-staged projection GEMMs
  proj_gemm_kernel<<<1024, 256, 0, stream>>>(x_bf, wkT, ktr, 1);
  proj_gemm_kernel<<<1024, 256, 0, stream>>>(x_bf, wvT, vmat, 0);

  const float SCALE = 0.0625f;  // 256^-0.5
  for (int it = 0; it < 3; ++it) {
    ln_rows_kernel<<<256, 256, 0, stream>>>(slots, ln_s_g, ln_s_b, s_bf);
    cvt_bf16_kernel<<<256, 256, 0, stream>>>(slots, h_bf, 65536);
    // q = LN(slots) @ Wq^T
    gemm(s_bf, wqT, q_bf, 16, 16, 256, 256, 256, 256,
         0, 0, 0, 1, nullptr, nullptr, 1.0f, F_OUTBF16);
    // dots[b,i,j] = SCALE * q . k   (A rows 8..15 are pad, dropped)
    gemm(q_bf, ktr, dots, 1, 64, 256, 256, 1024, 1024,
         2048, 262144, 8192, 32, nullptr, nullptr, SCALE, F_ROWS8);
    softmax_slots_kernel<<<128, 256, 0, stream>>>(dots, attn_f);
    attn_norm_kernel<<<256, 256, 0, stream>>>(attn_f, attn_b);
    // updates[b,i,d] = attn @ v
    gemm(attn_b, vmat, u_bf, 1, 16, 1024, 1024, 256, 256,
         8192, 262144, 2048, 32, nullptr, nullptr, 1.0f, F_OUTBF16 | F_ROWS8);
    // GRU gates
    gemm(u_bf, wihT, gi, 16, 48, 256, 256, 768, 768,
         0, 0, 0, 1, gru_bih, nullptr, 1.0f, 0);
    gemm(h_bf, whhT, gh, 16, 48, 256, 256, 768, 768,
         0, 0, 0, 1, gru_bhh, nullptr, 1.0f, 0);
    gru_ln_kernel<<<256, 256, 0, stream>>>(gi, gh, slots, ln_ff_g, ln_ff_b,
                                           smid, s_bf);
    // MLP
    gemm(s_bf, w1T, hmid, 16, 64, 256, 256, 1024, 1024,
         0, 0, 0, 1, mlp_b1, nullptr, 1.0f, F_OUTBF16 | F_RELU);
    gemm(hmid, w2T, slots, 16, 16, 1024, 1024, 256, 256,
         0, 0, 0, 1, mlp_b2, smid, 1.0f, 0);
  }

  copy_f32_kernel<<<256, 256, 0, stream>>>(slots, (float*)d_out, 65536);
}
